// SConv1_4423816315474
// MI455X (gfx1250) — matrix operand order repaired and checked by reading, not compiled, not run on hardware
//
#include <hip/hip_runtime.h>

// ---------------------------------------------------------------------------
// CDNA5 (gfx1250) GNN forward: WMMA bf16 GEMMs + L2 atomics for segment ops.
// wave32; each wave owns a 16-row x 128-col output chunk with the A stripe
// (<=8 bf16 fragments) held in VGPRs across all 8 N-tiles; activations are
// compile-time template params using v_exp_f32 / v_rcp_f32 fast paths.
// ---------------------------------------------------------------------------

typedef __attribute__((ext_vector_type(16))) __bf16 v16bf;
typedef __attribute__((ext_vector_type(8)))  float  v8f;

#define DEV __device__ __forceinline__

// ---- fast transcendentals (hardware v_exp_f32 / v_rcp_f32) ----
DEV float fast_sigmoid(float x) {
  return __builtin_amdgcn_rcpf(1.f + __expf(-x));
}
DEV float fast_tanh(float x) {
  // tanh(x) = 1 - 2/(exp(2x)+1)
  return 1.f - 2.f * __builtin_amdgcn_rcpf(1.f + __expf(2.f * x));
}

template <int ACT>
DEV float act_apply(float x) {
  if (ACT == 1) return fast_tanh(x);
  if (ACT == 2) return x > 0.f ? x : 0.1f * x;
  if (ACT == 3) return fast_sigmoid(x);
  return x;
}

// A fragment: 16x32 bf16 tile of X (rows m0..m0+15, cols k0..k0+31).
// Contiguous k order used here: lanes 0-15 -> K 0..15, lanes 16-31 -> K 16..31 (matches B).
// X may be a concat of two row-major blocks: cols [0,K0) from X0, [K0,K) from X1.
DEV v16bf load_a_frag(const float* __restrict__ X0, const float* __restrict__ X1,
                      int K0, int K, int m0, int k0, int lane) {
  const int row = m0 + (lane & 15);
  const int hs  = lane >> 4;
  v16bf a;
#pragma unroll
  for (int e = 0; e < 16; ++e) {
    const int kk = hs * 16 + e;
    const int k  = k0 + kk;
    float val;
    if (k < K0) val = X0[(long)row * K0 + k];
    else        val = X1[(long)row * (K - K0) + (k - K0)];
    a[e] = (__bf16)val;
  }
  return a;
}

// B fragment: 32x16 bf16 tile of W^T: W[Nout,K] row-major, cols n0..n0+15,
// K rows k0..k0+31. Lanes 0-15 hold K 0..15, lanes 16-31 hold K 16..31.
DEV v16bf load_b_frag(const float* __restrict__ W, int K, int n0, int k0, int lane) {
  const int col  = n0 + (lane & 15);
  const int base = k0 + (lane >> 4) * 16;
  v16bf b;
#pragma unroll
  for (int e = 0; e < 16; ++e)
    b[e] = (__bf16)W[(long)col * K + base + e];
  return b;
}

DEV v8f wmma_bf16(v16bf a, v16bf b, v8f acc) {
  return __builtin_amdgcn_wmma_f32_16x16x32_bf16(false, a, false, b, (short)0, acc,
                                                 false, false);
}

// ---------------------------------------------------------------------------
// Generic GEMM: Out[M,Nout] = act(X[M,K] @ W[Nout,K]^T + bias), X = cat(X0,X1).
// One wave per 16-row x (<=128-col) output chunk; A stripe register-resident
// when K <= 256 (all large GEMMs here: K = 128 / 256 / 384-out).
// ---------------------------------------------------------------------------
template <int ACT>
__global__ void gemm_bias_act_kernel(const float* __restrict__ X0,
                                     const float* __restrict__ X1, int K0,
                                     const float* __restrict__ W,
                                     const float* __restrict__ bias,
                                     float* __restrict__ Out,
                                     int M, int K, int Nout) {
  const int lane   = threadIdx.x & 31;
  const int wave   = blockIdx.x * (blockDim.x >> 5) + (threadIdx.x >> 5);
  const int stride = gridDim.x * (blockDim.x >> 5);
  const int nt      = Nout >> 4;
  const int nchunks = (nt + 7) >> 3;             // 8 n-tiles (128 cols) per chunk
  const long total  = (long)(M >> 4) * nchunks;
  const int kf = K >> 5;                         // K fragments of 32

  for (long t = wave; t < total; t += stride) {
    const int m0 = (int)(t / nchunks) << 4;
    const int nc = (int)(t % nchunks);
    const int nbeg = nc << 3;
    const int nend = (nbeg + 8 < nt) ? nbeg + 8 : nt;

    v16bf afrag[8];
    const bool areg = (kf <= 8);
    if (areg) {
      for (int i = 0; i < kf; ++i)
        afrag[i] = load_a_frag(X0, X1, K0, K, m0, i * 32, lane);
    }

    for (int ntile = nbeg; ntile < nend; ++ntile) {
      const int n0 = ntile << 4;
      v8f acc = {};
      if (areg) {
        for (int i = 0; i < kf; ++i)
          acc = wmma_bf16(afrag[i], load_b_frag(W, K, n0, i * 32, lane), acc);
      } else {                                   // K=512 path (tiny m2s GEMM)
        for (int k0 = 0; k0 < K; k0 += 32)
          acc = wmma_bf16(load_a_frag(X0, X1, K0, K, m0, k0, lane),
                          load_b_frag(W, K, n0, k0, lane), acc);
      }
      const int col = n0 + (lane & 15);
      const float bv = bias ? bias[col] : 0.f;
#pragma unroll
      for (int j = 0; j < 8; ++j) {
        const int row = m0 + (lane >> 4) * 8 + j;
        Out[(long)row * Nout + col] = act_apply<ACT>(acc[j] + bv);
      }
    }
  }
}

// ---------------------------------------------------------------------------
// Attention logits, one head:
//   logits[n] = sum_h tanh(v@hA^T+b)[n,h] * dsup[gid[n],h] * hC[h] + hC_b
// A stripe (v rows, K=128 -> 4 frags) register-resident; shfl_xor col-reduce.
// ---------------------------------------------------------------------------
__global__ void attn_logits_kernel(const float* __restrict__ v,
                                   const float* __restrict__ hA_w,   // [128,128] head slice
                                   const float* __restrict__ hA_b,   // [128]
                                   const float* __restrict__ hC_w,   // [128]
                                   const float* __restrict__ hC_bp,  // [KH]
                                   const float* __restrict__ dsup,   // [B,128] head slice
                                   const int* __restrict__ gid,
                                   float* __restrict__ logits,       // [N] head slice
                                   int N, int head) {
  const int lane   = threadIdx.x & 31;
  const int wave   = blockIdx.x * (blockDim.x >> 5) + (threadIdx.x >> 5);
  const int stride = gridDim.x * (blockDim.x >> 5);
  const int tiles  = N >> 4;
  const float cb = hC_bp[head];
  for (int t = wave; t < tiles; t += stride) {
    const int m0 = t << 4;
    v16bf afrag[4];
#pragma unroll
    for (int i = 0; i < 4; ++i)
      afrag[i] = load_a_frag(v, nullptr, 128, 128, m0, i * 32, lane);

    float partial[8];
    int   g[8];
#pragma unroll
    for (int j = 0; j < 8; ++j) {
      partial[j] = 0.f;
      g[j] = gid[m0 + (lane >> 4) * 8 + j];
    }
    for (int n0 = 0; n0 < 128; n0 += 16) {
      v8f acc = {};
#pragma unroll
      for (int i = 0; i < 4; ++i)
        acc = wmma_bf16(afrag[i], load_b_frag(hA_w, 128, n0, i * 32, lane), acc);
      const int col = n0 + (lane & 15);
      const float bb = hA_b[col], cc = hC_w[col];
#pragma unroll
      for (int j = 0; j < 8; ++j)
        partial[j] += fast_tanh(acc[j] + bb) * dsup[(long)g[j] * 128 + col] * cc;
    }
#pragma unroll
    for (int j = 0; j < 8; ++j) {
      float p = partial[j];
      p += __shfl_xor(p, 1, 32);
      p += __shfl_xor(p, 2, 32);
      p += __shfl_xor(p, 4, 32);
      p += __shfl_xor(p, 8, 32);
      if ((lane & 15) == 0)
        logits[m0 + (lane >> 4) * 8 + j] = p + cb;
    }
  }
}

// ----- segment softmax via encoded-uint atomicMax + f32 atomic add ---------
DEV unsigned enc_f32(float f) {
  int i = __float_as_int(f);
  return (unsigned)(i ^ ((i >> 31) | 0x80000000));
}
DEV float dec_u32(unsigned u) {
  int i = (u & 0x80000000u) ? (int)(u ^ 0x80000000u) : ~(int)u;
  return __int_as_float(i);
}

__global__ void softmax_init_kernel(unsigned* __restrict__ gmax,
                                    float* __restrict__ den, int n) {
  for (int i = blockIdx.x * blockDim.x + threadIdx.x; i < n;
       i += gridDim.x * blockDim.x) {
    gmax[i] = 0x007FFFFFu;  // encoded -inf
    den[i]  = 0.f;
  }
}

__global__ void seg_max_kernel(const float* __restrict__ logits,
                               const int* __restrict__ gid,
                               unsigned* __restrict__ gmax, int N, int B, int KH) {
  const long total = (long)KH * N;
  for (long i = blockIdx.x * blockDim.x + threadIdx.x; i < total;
       i += (long)gridDim.x * blockDim.x) {
    const int head = (int)(i / N), n = (int)(i % N);
    atomicMax(&gmax[head * B + gid[n]], enc_f32(logits[i]));
  }
}

__global__ void seg_expsum_kernel(float* __restrict__ logits,
                                  const int* __restrict__ gid,
                                  const unsigned* __restrict__ gmax,
                                  float* __restrict__ den, int N, int B, int KH) {
  const long total = (long)KH * N;
  for (long i = blockIdx.x * blockDim.x + threadIdx.x; i < total;
       i += (long)gridDim.x * blockDim.x) {
    const int head = (int)(i / N), n = (int)(i % N);
    const float mx = dec_u32(gmax[head * B + gid[n]]);
    const float ex = __expf(logits[i] - mx);
    logits[i] = ex;
    atomicAdd(&den[head * B + gid[n]], ex);
  }
}

// ---------------------------------------------------------------------------
// hv pooling, one head: m2s_raw[g, head*128+h] += (v@hD^T + b)[n,h] * att[n]
// att folded as ex/den; segment-sum via L2 f32 atomics; A stripe in VGPRs.
// ---------------------------------------------------------------------------
__global__ void attn_pool_kernel(const float* __restrict__ v,
                                 const float* __restrict__ hD_w,  // head slice [128,128]
                                 const float* __restrict__ hD_b,  // [128]
                                 const float* __restrict__ ex,    // [N] head slice
                                 const float* __restrict__ den,   // [B] head slice
                                 const int* __restrict__ gid,
                                 float* __restrict__ m2s_raw,     // [B, 4*128]
                                 int N, int head) {
  const int lane   = threadIdx.x & 31;
  const int wave   = blockIdx.x * (blockDim.x >> 5) + (threadIdx.x >> 5);
  const int stride = gridDim.x * (blockDim.x >> 5);
  const int tiles  = N >> 4;
  for (int t = wave; t < tiles; t += stride) {
    const int m0 = t << 4;
    v16bf afrag[4];
#pragma unroll
    for (int i = 0; i < 4; ++i)
      afrag[i] = load_a_frag(v, nullptr, 128, 128, m0, i * 32, lane);

    float att[8];
    int   g[8];
#pragma unroll
    for (int j = 0; j < 8; ++j) {
      const int row = m0 + (lane >> 4) * 8 + j;
      g[j]   = gid[row];
      att[j] = ex[row] * __builtin_amdgcn_rcpf(den[g[j]]);
    }
    for (int n0 = 0; n0 < 128; n0 += 16) {
      v8f acc = {};
#pragma unroll
      for (int i = 0; i < 4; ++i)
        acc = wmma_bf16(afrag[i], load_b_frag(hD_w, 128, n0, i * 32, lane), acc);
      const int col = n0 + (lane & 15);
      const float bb = hD_b[col];
#pragma unroll
      for (int j = 0; j < 8; ++j)
        atomicAdd(&m2s_raw[(long)g[j] * 512 + head * 128 + col],
                  (acc[j] + bb) * att[j]);
    }
  }
}

// ---------------------------------------------------------------------------
// Edge messages: ve = lrelu((e @ K_w^T + K_b) * v[src]); sve[dst] += ve.
// K=16 zero-padded to 32 -> one bf16 WMMA per 16-edge x 16-out tile.
// v gather + sve scatter both hit the 192MB L2 (v is only 20MB).
// ---------------------------------------------------------------------------
__global__ void edge_msg_kernel(const float* __restrict__ efeat,  // [E,16]
                                const float* __restrict__ Kw,     // [128,16]
                                const float* __restrict__ Kb,     // [128]
                                const float* __restrict__ v,      // [N,128]
                                const int* __restrict__ esrc,
                                const int* __restrict__ edst,
                                float* __restrict__ sve, int E) {
  const int lane   = threadIdx.x & 31;
  const int wave   = blockIdx.x * (blockDim.x >> 5) + (threadIdx.x >> 5);
  const int stride = gridDim.x * (blockDim.x >> 5);
  const int tiles  = E >> 4;
  for (int t = wave; t < tiles; t += stride) {
    const int m0 = t << 4;
    int src[8], dst[8];
#pragma unroll
    for (int j = 0; j < 8; ++j) {
      const int r = m0 + (lane >> 4) * 8 + j;
      src[j] = esrc[r];
      dst[j] = edst[r];
    }
    // A fragment, K padded 16 -> 32 with zeros
    const int rowA = m0 + (lane & 15);
    const int hs = lane >> 4;
    v16bf a;
#pragma unroll
    for (int q = 0; q < 16; ++q) {
      const int kk = hs * 16 + q;
      a[q] = (__bf16)((kk < 16) ? efeat[(long)rowA * 16 + kk] : 0.f);
    }
    for (int n0 = 0; n0 < 128; n0 += 16) {
      const int col = n0 + (lane & 15);
      const int kbase = (lane >> 4) * 16;
      v16bf b;
#pragma unroll
      for (int q = 0; q < 16; ++q) {
        const int kk = kbase + q;
        b[q] = (__bf16)((kk < 16) ? Kw[(long)col * 16 + kk] : 0.f);
      }
      v8f acc = {};
      acc = wmma_bf16(a, b, acc);
      const float kb = Kb[col];
#pragma unroll
      for (int j = 0; j < 8; ++j) {
        float val = (acc[j] + kb) * v[(long)src[j] * 128 + col];
        val = (val > 0.f) ? val : 0.1f * val;
        atomicAdd(&sve[(long)dst[j] * 128 + col], val);
      }
    }
  }
}

// ---------------------------------------------------------------------------
// Elementwise: z = sigmoid(zA + zB[g]); h = z * bvec[g] + (1-z) * a  (in-place ok)
// gidp == nullptr -> g = row (super-node path).
// ---------------------------------------------------------------------------
__global__ void gate_h_kernel(const float* __restrict__ zA,
                              const float* __restrict__ zB,
                              const float* __restrict__ bvec,
                              const float* __restrict__ a,
                              const int* __restrict__ gidp,
                              float* __restrict__ h, int M, int Dd) {
  const long total = (long)M * Dd;
  for (long i = blockIdx.x * blockDim.x + threadIdx.x; i < total;
       i += (long)gridDim.x * blockDim.x) {
    const int row = (int)(i / Dd), d = (int)(i % Dd);
    const int g = gidp ? gidp[row] : row;
    const float z = fast_sigmoid(zA[i] + zB[(long)g * Dd + d]);
    h[i] = z * bvec[(long)g * Dd + d] + (1.f - z) * a[i];
  }
}

// GRU elementwise combine (gates r,z,n): out = (1-z)*n + z*h
__global__ void gru_combine_kernel(const float* __restrict__ gi,  // [M,3D]
                                   const float* __restrict__ gh,  // [M,3D]
                                   const float* __restrict__ h,   // [M,D]
                                   float* __restrict__ out, int M, int Dd) {
  const long total = (long)M * Dd;
  for (long i = blockIdx.x * blockDim.x + threadIdx.x; i < total;
       i += (long)gridDim.x * blockDim.x) {
    const int row = (int)(i / Dd), d = (int)(i % Dd);
    const long b3 = (long)row * 3 * Dd;
    const float r = fast_sigmoid(gi[b3 + d] + gh[b3 + d]);
    const float z = fast_sigmoid(gi[b3 + Dd + d] + gh[b3 + Dd + d]);
    const float n = fast_tanh(gi[b3 + 2 * Dd + d] + r * gh[b3 + 2 * Dd + d]);
    out[i] = (1.f - z) * n + z * h[i];
  }
}

__global__ void zero_kernel(float* __restrict__ p, long n) {
  for (long i = blockIdx.x * blockDim.x + threadIdx.x; i < n;
       i += (long)gridDim.x * blockDim.x)
    p[i] = 0.f;
}

// ---------------------------------------------------------------------------
// Host-side orchestration
// ---------------------------------------------------------------------------
static inline int tile_grid(long tiles) {
  long b = (tiles + 7) / 8;       // 8 waves (256 threads) per block
  if (b > 8192) b = 8192;
  if (b < 1) b = 1;
  return (int)b;
}
static inline int ew_grid(long n) {
  long b = (n + 255) / 256;
  if (b > 4096) b = 4096;
  if (b < 1) b = 1;
  return (int)b;
}

extern "C" void kernel_launch(void* const* d_in, const int* in_sizes, int n_in,
                              void* d_out, int out_size, void* d_ws, size_t ws_size,
                              hipStream_t stream) {
  (void)n_in; (void)out_size; (void)ws_size;
  const int D = 128, EDIM = 16, KH = 4;
  const int N = in_sizes[0] / D;      // 40000
  const int E = in_sizes[1] / EDIM;   // 640000
  const int B = in_sizes[2] / D;      // 64

  const float* v = (const float*)d_in[0];
  const float* e = (const float*)d_in[1];
  const float* s = (const float*)d_in[2];
  // params in dict order starting at index 3
  const float* A_w  = (const float*)d_in[3];
  const float* A_b  = (const float*)d_in[4];
  const float* hA_w = (const float*)d_in[5];
  const float* hA_b = (const float*)d_in[6];
  const float* hB_w = (const float*)d_in[7];
  const float* hB_b = (const float*)d_in[8];
  const float* hC_w = (const float*)d_in[9];
  const float* hC_b = (const float*)d_in[10];
  const float* hD_w = (const float*)d_in[11];
  const float* hD_b = (const float*)d_in[12];
  const float* B_w  = (const float*)d_in[13];
  const float* B_b  = (const float*)d_in[14];
  const float* C_w  = (const float*)d_in[15];
  const float* C_b  = (const float*)d_in[16];
  const float* E_w  = (const float*)d_in[17];
  const float* E_b  = (const float*)d_in[18];
  const float* K_w  = (const float*)d_in[19];
  const float* K_b  = (const float*)d_in[20];
  const float* gm_A_w = (const float*)d_in[21];
  const float* gm_A_b = (const float*)d_in[22];
  const float* gm_B_w = (const float*)d_in[23];
  const float* gm_B_b = (const float*)d_in[24];
  const float* gm_w_ih = (const float*)d_in[25];
  const float* gm_b_ih = (const float*)d_in[26];
  const float* gm_w_hh = (const float*)d_in[27];
  const float* gm_b_hh = (const float*)d_in[28];
  const float* gs_A_w = (const float*)d_in[29];
  const float* gs_A_b = (const float*)d_in[30];
  const float* gs_B_w = (const float*)d_in[31];
  const float* gs_B_b = (const float*)d_in[32];
  const float* gs_w_ih = (const float*)d_in[33];
  const float* gs_b_ih = (const float*)d_in[34];
  const float* gs_w_hh = (const float*)d_in[35];
  const float* gs_b_hh = (const float*)d_in[36];
  const int* edge_src = (const int*)d_in[37];
  const int* edge_dst = (const int*)d_in[38];
  const int* gid      = (const int*)d_in[39];

  float* out = (float*)d_out;        // vv [N,128] then ss [B,128]

  // ---- workspace carve-up (floats) ----
  float* ws = (float*)d_ws;
  long off = 0;
  float* s2s     = ws + off; off += (long)B * D;
  float* sC      = ws + off; off += (long)B * D;
  float* dsup    = ws + off; off += (long)KH * B * D;
  float* m2s_raw = ws + off; off += (long)B * KH * D;   // [B, 512]
  float* m2s     = ws + off; off += (long)B * D;
  float* logits  = ws + off; off += (long)KH * N;
  unsigned* gmax = (unsigned*)(ws + off); off += KH * B;
  float* den     = ws + off; off += KH * B;
  float* sve     = ws + off; off += (long)N * D;        // reused as zA later
  float* m2m     = ws + off; off += (long)N * D;        // becomes h (in-place)
  float* zB      = ws + off; off += (long)B * D;
  float* gi      = ws + off; off += (long)N * 3 * D;
  float* gh      = ws + off; off += (long)N * 3 * D;
  float* zAs     = ws + off; off += (long)B * D;
  float* zBs     = ws + off; off += (long)B * D;
  float* hs      = ws + off; off += (long)B * D;
  float* gis     = ws + off; off += (long)B * 3 * D;
  float* ghs     = ws + off; off += (long)B * 3 * D;
  float* zA = sve;  // alias: sve dead once m2m is computed

  const dim3 blk(256);

  // ---- init accumulators ----
  zero_kernel<<<ew_grid((long)N * D), blk, 0, stream>>>(sve, (long)N * D);
  zero_kernel<<<ew_grid((long)B * KH * D), blk, 0, stream>>>(m2s_raw, (long)B * KH * D);
  softmax_init_kernel<<<1, 256, 0, stream>>>(gmax, den, KH * B);

  // ---- super-node small GEMMs (M=64) ----
  gemm_bias_act_kernel<1><<<tile_grid(B / 16), blk, 0, stream>>>(
      s, nullptr, D, A_w, A_b, s2s, B, D, D);
  gemm_bias_act_kernel<1><<<tile_grid(B / 16), blk, 0, stream>>>(
      s, nullptr, D, C_w, C_b, sC, B, D, D);
  for (int k = 0; k < KH; ++k)
    gemm_bias_act_kernel<1><<<tile_grid(B / 16), blk, 0, stream>>>(
        s, nullptr, D, hB_w + (long)k * D * D, hB_b + k * D,
        dsup + (long)k * B * D, B, D, D);

  // ---- attention logits + segment softmax ----
  for (int k = 0; k < KH; ++k)
    attn_logits_kernel<<<tile_grid(N / 16), blk, 0, stream>>>(
        v, hA_w + (long)k * D * D, hA_b + k * D, hC_w + k * D, hC_b,
        dsup + (long)k * B * D, gid, logits + (long)k * N, N, k);
  seg_max_kernel<<<ew_grid((long)KH * N), blk, 0, stream>>>(logits, gid, gmax, N, B, KH);
  seg_expsum_kernel<<<ew_grid((long)KH * N), blk, 0, stream>>>(logits, gid, gmax, den, N, B, KH);

  // ---- attention pooling -> m2s_raw [B,512], then m2s = tanh(GEMM K=512) ----
  for (int k = 0; k < KH; ++k)
    attn_pool_kernel<<<tile_grid(N / 16), blk, 0, stream>>>(
        v, hD_w + (long)k * D * D, hD_b + k * D, logits + (long)k * N,
        den + k * B, gid, m2s_raw, N, k);
  gemm_bias_act_kernel<1><<<tile_grid(B / 16), blk, 0, stream>>>(
      m2s_raw, nullptr, KH * D, B_w, B_b, m2s, B, KH * D, D);  // K=512 stream path

  // ---- edge-gated message passing ----
  edge_msg_kernel<<<tile_grid(E / 16), blk, 0, stream>>>(
      e, K_w, K_b, v, edge_src, edge_dst, sve, E);
  gemm_bias_act_kernel<2><<<tile_grid(N / 16), blk, 0, stream>>>(
      sve, v, D, E_w, E_b, m2m, N, 2 * D, D);      // X = cat(sve, v), lrelu

  // ---- gm gate + GRU (node update) ----
  gemm_bias_act_kernel<0><<<tile_grid(N / 16), blk, 0, stream>>>(
      m2m, nullptr, D, gm_A_w, gm_A_b, zA, N, D, D);
  gemm_bias_act_kernel<0><<<tile_grid(B / 16), blk, 0, stream>>>(
      sC, nullptr, D, gm_B_w, gm_B_b, zB, B, D, D);
  gate_h_kernel<<<ew_grid((long)N * D), blk, 0, stream>>>(
      zA, zB, sC, m2m, gid, m2m, N, D);            // h in-place over m2m
  gemm_bias_act_kernel<0><<<tile_grid((long)(N / 16) * 3), blk, 0, stream>>>(
      v, nullptr, D, gm_w_ih, gm_b_ih, gi, N, D, 3 * D);
  gemm_bias_act_kernel<0><<<tile_grid((long)(N / 16) * 3), blk, 0, stream>>>(
      m2m, nullptr, D, gm_w_hh, gm_b_hh, gh, N, D, 3 * D);
  gru_combine_kernel<<<ew_grid((long)N * D), blk, 0, stream>>>(
      gi, gh, m2m, out, N, D);                     // vv -> d_out[0 : N*D]

  // ---- gs gate + GRU (super-node update) ----
  gemm_bias_act_kernel<0><<<tile_grid(B / 16), blk, 0, stream>>>(
      s2s, nullptr, D, gs_A_w, gs_A_b, zAs, B, D, D);
  gemm_bias_act_kernel<0><<<tile_grid(B / 16), blk, 0, stream>>>(
      m2s, nullptr, D, gs_B_w, gs_B_b, zBs, B, D, D);
  gate_h_kernel<<<ew_grid((long)B * D), blk, 0, stream>>>(
      zAs, zBs, m2s, s2s, nullptr, hs, B, D);
  gemm_bias_act_kernel<0><<<tile_grid((B / 16) * 3), blk, 0, stream>>>(
      s, nullptr, D, gs_w_ih, gs_b_ih, gis, B, D, 3 * D);
  gemm_bias_act_kernel<0><<<tile_grid((B / 16) * 3), blk, 0, stream>>>(
      hs, nullptr, D, gs_w_hh, gs_b_hh, ghs, B, D, 3 * D);
  gru_combine_kernel<<<ew_grid((long)B * D), blk, 0, stream>>>(
      gis, ghs, hs, out + (long)N * D, B, D);      // ss -> d_out[N*D : ]
}
